// SwinTransformerBlock3D_7447473291276
// MI455X (gfx1250) — matrix operand order repaired and checked by reading, not compiled, not run on hardware
//
#include <hip/hip_runtime.h>
#include <hip/hip_bf16.h>
#include <math.h>

// ---------------- types ----------------
typedef __attribute__((ext_vector_type(16))) _Float16 v16h;
typedef __attribute__((ext_vector_type(8)))  _Float16 v8h;
typedef __attribute__((ext_vector_type(8)))  float    v8f;

// ---------------- problem constants ----------------
#define DD    16
#define HH    56
#define WW    56
#define CCH   256
#define HID   1024
#define NHEAD 8
#define HDIM  32          // head dim
#define WIN_D 8
#define WIN_H 7
#define WIN_W 7
#define NTOK  392         // tokens per window = 8*7*7
#define NPAD  416         // padded tokens: 416 = 26*16 = 13*32
#define NWIN  128         // (16/8)*(56/7)*(56/7)
#define TOKS  50176       // DD*HH*WW
#define SSD   4
#define SSH   3
#define SSW   3
#define QSCALE 0.17677669529663687f  // 32^-0.5

// ---------------- WMMA fragment loaders (contiguous k order, same for A and B) ----
// A matrix 16x32 f16: lane row = lane&15; k = hi*16 + e (16 contiguous halves)
__device__ __forceinline__ v16h load_afrag(const _Float16* __restrict__ A, int lda,
                                           int row0, int k0, int rA, int hi) {
  return *(const v16h*)(A + (size_t)(row0 + rA) * lda + k0 + hi * 16);
}
// B matrix 32x16 f16: lane col = lane&15; k = hi*16 + e (16 contiguous halves)
// W is stored row-major (Ncols x K), i.e. B^T, so B(k, n) = W[n, k]
__device__ __forceinline__ v16h load_bfrag(const _Float16* __restrict__ W, int ldw,
                                           int col0, int k0, int nB, int hi) {
  return *(const v16h*)(W + (size_t)(col0 + nB) * ldw + k0 + hi * 16);
}

// 64x64 register-blocked wave GEMM core: 16 WMMAs per 32-deep K step,
// 4 A-frags x 4 B-frags -> 32 FLOP/byte from cache (vs 8 for 16x16 tiles).
// Single prefetch of the wave's streaming A slab before the branch-free K loop.
__device__ __forceinline__ void gemm_core64(const _Float16* __restrict__ A, int lda,
                                            const _Float16* __restrict__ W, int ldw,
                                            int row0, int col0, int K,
                                            int rA, int hi, v8f acc[4][4]) {
#pragma unroll
  for (int mb = 0; mb < 4; mb++)
#pragma unroll
    for (int nb = 0; nb < 4; nb++) acc[mb][nb] = (v8f){};
  __builtin_prefetch(A + (size_t)(row0 + rA) * lda, 0, 0);  // global_prefetch_b8
  for (int k0 = 0; k0 < K; k0 += 32) {
    v16h a[4];
#pragma unroll
    for (int mb = 0; mb < 4; mb++) a[mb] = load_afrag(A, lda, row0 + mb * 16, k0, rA, hi);
#pragma unroll
    for (int nb = 0; nb < 4; nb++) {
      v16h b = load_bfrag(W, ldw, col0 + nb * 16, k0, rA, hi);
#pragma unroll
      for (int mb = 0; mb < 4; mb++)
        acc[mb][nb] = __builtin_amdgcn_wmma_f32_16x16x32_f16(false, a[mb], false, b,
                                                             (short)0, acc[mb][nb], false, false);
    }
  }
}

// ---------------- helpers ----------------
__global__ __launch_bounds__(256)
void f32_to_f16_kernel(const float* __restrict__ s, _Float16* __restrict__ d, int n) {
  int i = blockIdx.x * 256 + threadIdx.x;
  if (i < n) d[i] = (_Float16)s[i];
}

__global__ __launch_bounds__(256)
void zero_kernel(unsigned int* __restrict__ p, size_t nwords) {
  size_t i = (size_t)blockIdx.x * 256 + threadIdx.x;
  size_t stride = (size_t)gridDim.x * 256;
  for (; i < nwords; i += stride) p[i] = 0u;
}

// ---------------- LN1 + cyclic shift + window partition -> win f16 ------------
__global__ __launch_bounds__(256)
void ln1_shift_part_kernel(const float* __restrict__ x, const float* __restrict__ g,
                           const float* __restrict__ b, _Float16* __restrict__ win) {
  int wave = threadIdx.x >> 5, lane = threadIdx.x & 31;
  size_t t = (size_t)blockIdx.x * 8 + wave;
  if (t >= (size_t)TOKS) return;
  int w = (int)(t / NTOK), nn = (int)(t % NTOK);
  int dbl = w >> 6, hbl = (w >> 3) & 7, wbl = w & 7;
  int td = nn / 49, th = (nn / 7) % 7, tw = nn % 7;
  int sd = dbl * WIN_D + td, sh = hbl * WIN_H + th, sw = wbl * WIN_W + tw;
  int id = (sd + SSD) & 15;
  int ih = sh + SSH; if (ih >= HH) ih -= HH;
  int iw = sw + SSW; if (iw >= WW) iw -= WW;
  size_t src = ((size_t)(id * HH + ih) * WW + iw) * CCH;
  const float4* p = (const float4*)(x + src + lane * 8);
  float4 u0 = p[0], u1 = p[1];
  float v[8] = {u0.x,u0.y,u0.z,u0.w,u1.x,u1.y,u1.z,u1.w};
  float s = 0.f, q = 0.f;
#pragma unroll
  for (int i = 0; i < 8; i++) { s += v[i]; q += v[i] * v[i]; }
  for (int off = 16; off >= 1; off >>= 1) { s += __shfl_xor(s, off); q += __shfl_xor(q, off); }
  float mean = s * (1.0f / CCH);
  float var  = q * (1.0f / CCH) - mean * mean;
  float inv  = rsqrtf(var + 1e-5f);
  _Float16* o = win + t * CCH + lane * 8;
#pragma unroll
  for (int i = 0; i < 8; i++) {
    int ch = lane * 8 + i;
    o[i] = (_Float16)((v[i] - mean) * inv * g[ch] + b[ch]);
  }
}

// ---------------- LN2 (image order) -> h2 f16 ---------------------------------
__global__ __launch_bounds__(256)
void ln2_kernel(const float* __restrict__ x2, const float* __restrict__ g,
                const float* __restrict__ b, _Float16* __restrict__ h2) {
  int wave = threadIdx.x >> 5, lane = threadIdx.x & 31;
  size_t t = (size_t)blockIdx.x * 8 + wave;
  if (t >= (size_t)TOKS) return;
  const float4* p = (const float4*)(x2 + t * CCH + lane * 8);
  float4 u0 = p[0], u1 = p[1];
  float v[8] = {u0.x,u0.y,u0.z,u0.w,u1.x,u1.y,u1.z,u1.w};
  float s = 0.f, q = 0.f;
#pragma unroll
  for (int i = 0; i < 8; i++) { s += v[i]; q += v[i] * v[i]; }
  for (int off = 16; off >= 1; off >>= 1) { s += __shfl_xor(s, off); q += __shfl_xor(q, off); }
  float mean = s * (1.0f / CCH);
  float var  = q * (1.0f / CCH) - mean * mean;
  float inv  = rsqrtf(var + 1e-5f);
  _Float16* o = h2 + t * CCH + lane * 8;
#pragma unroll
  for (int i = 0; i < 8; i++) {
    int ch = lane * 8 + i;
    o[i] = (_Float16)((v[i] - mean) * inv * g[ch] + b[ch]);
  }
}

// ---------------- QKV GEMM: win(50176x256) x qkv_w^T(256x768) -----------------
__global__ __launch_bounds__(256)
void qkv_gemm_kernel(const _Float16* __restrict__ A, const _Float16* __restrict__ Wt,
                     const float* __restrict__ bias, _Float16* __restrict__ Qb,
                     _Float16* __restrict__ Kb, _Float16* __restrict__ Vt) {
  const int NB = 768 / 64;
  int lane = threadIdx.x & 31, wid = threadIdx.x >> 5;
  long tile = (long)blockIdx.x * 8 + wid;
  if (tile >= (long)(TOKS / 64) * NB) return;
  int nt = (int)(tile % NB), mt = (int)(tile / NB);
  int rA = lane & 15, hi = lane >> 4;
  int row0 = mt * 64, col0 = nt * 64;
  v8f acc[4][4];
  gemm_core64(A, CCH, Wt, CCH, row0, col0, CCH, rA, hi, acc);
#pragma unroll
  for (int nb = 0; nb < 4; nb++) {
    int jj = col0 + nb * 16 + rA;
    float bj = bias[jj];
    int sel = jj >> 8, head = (jj >> 5) & 7, dch = jj & 31;
#pragma unroll
    for (int mb = 0; mb < 4; mb++) {
#pragma unroll
      for (int r = 0; r < 8; r++) {
        int m = row0 + mb * 16 + r + hi * 8;
        int w = m / NTOK, nn = m % NTOK;
        float val = acc[mb][nb][r] + bj;
        size_t wh = (size_t)(w * NHEAD + head);
        if (sel == 0)      Qb[(wh * NPAD + nn) * HDIM + dch] = (_Float16)(val * QSCALE);
        else if (sel == 1) Kb[(wh * NPAD + nn) * HDIM + dch] = (_Float16)val;
        else               Vt[(wh * HDIM + dch) * NPAD + nn] = (_Float16)val;
      }
    }
  }
}

// ---------------- fused attention: 1 wave per (window, head, 16-q tile) -------
__global__ __launch_bounds__(32)
void attn_kernel(const _Float16* __restrict__ Qb, const _Float16* __restrict__ Kb,
                 const _Float16* __restrict__ Vt, const float* __restrict__ rel_bias,
                 _Float16* __restrict__ O) {
  __shared__ __align__(16) float    sc[16][NPAD];
  __shared__ __align__(32) _Float16 at[16][NPAD];
  int lane = threadIdx.x;
  int rA = lane & 15, hi = lane >> 4;
  int qt0 = blockIdx.x * 16;
  int head = blockIdx.y;
  int w = blockIdx.z;
  int dbl = w >> 6, hbl = (w >> 3) & 7, wbl = w & 7;
  size_t whrow = (size_t)(w * NHEAD + head) * NPAD;

  // Q fragment (whole head dim K=32 fits one WMMA)
  v16h qa = load_afrag(Qb + whrow * HDIM, HDIM, qt0, 0, rA, hi);

  // query-side coords & shift-mask region per owned C row
  int qtd[8], qth[8], qtw[8], cq[8];
#pragma unroll
  for (int r = 0; r < 8; r++) {
    int qt = qt0 + r + hi * 8; if (qt >= NTOK) qt = NTOK - 1;
    int td = qt / 49, th = (qt / 7) % 7, tw = qt % 7;
    qtd[r] = td; qth[r] = th; qtw[r] = tw;
    int sd = dbl * WIN_D + td, sh = hbl * WIN_H + th, sw = wbl * WIN_W + tw;
    int rd = sd < 8 ? 0 : (sd < 12 ? 1 : 2);
    int rh = sh < 49 ? 0 : (sh < 53 ? 1 : 2);
    int rw = sw < 49 ? 0 : (sw < 53 ? 1 : 2);
    cq[r] = rd * 9 + rh * 3 + rw;
  }

  // scores = Q K^T + rel_bias + shift mask
  for (int j = 0; j < NPAD / 16; j++) {
    int kt = j * 16 + rA;
    v16h kb = *(const v16h*)(Kb + (whrow + j * 16 + rA) * HDIM + hi * 16);
    v8f c = {};
    c = __builtin_amdgcn_wmma_f32_16x16x32_f16(false, qa, false, kb, (short)0, c, false, false);
    bool valid = kt < NTOK;
    int ktd = 0, kth = 0, ktw = 0, ck = 0;
    if (valid) {
      ktd = kt / 49; kth = (kt / 7) % 7; ktw = kt % 7;
      int sd = dbl * WIN_D + ktd, sh = hbl * WIN_H + kth, sw = wbl * WIN_W + ktw;
      int rd = sd < 8 ? 0 : (sd < 12 ? 1 : 2);
      int rh = sh < 49 ? 0 : (sh < 53 ? 1 : 2);
      int rw = sw < 49 ? 0 : (sw < 53 ? 1 : 2);
      ck = rd * 9 + rh * 3 + rw;
    }
#pragma unroll
    for (int r = 0; r < 8; r++) {
      int m = r + hi * 8;
      float s;
      if (valid) {
        int idx = (qtd[r] - ktd + 7) * 169 + (qth[r] - kth + 6) * 13 + (qtw[r] - ktw + 6);
        float bias = rel_bias[idx * NHEAD + head];
        float mask = (cq[r] == ck) ? 0.f : -100.f;
        s = c[r] + bias + mask;
      } else s = -1e30f;
      sc[m][j * 16 + rA] = s;
    }
  }
  __syncthreads();

  // row softmax: all 32 lanes active; lane handles half a row, pair combines
  // via __shfl_xor(.,16). pad cols hold -1e30 -> exp underflows to 0.
  {
    const int half = NPAD / 2;           // 208
    int row = rA, c0 = hi * half;
    float mx = -1e30f;
    for (int cN = 0; cN < half; cN++) mx = fmaxf(mx, sc[row][c0 + cN]);
    mx = fmaxf(mx, __shfl_xor(mx, 16));
    float sum = 0.f;
    for (int cN = 0; cN < half; cN++) {
      float e = __expf(sc[row][c0 + cN] - mx);
      sc[row][c0 + cN] = e; sum += e;
    }
    sum += __shfl_xor(sum, 16);
    float inv = 1.0f / sum;
    for (int cN = 0; cN < half; cN++) at[row][c0 + cN] = (_Float16)(sc[row][c0 + cN] * inv);
  }
  __syncthreads();

  // out = attn * V  (two 16-col tiles cover d=32)
  v8f o0 = {}, o1 = {};
  size_t vrow = (size_t)(w * NHEAD + head) * HDIM;
  for (int k0 = 0; k0 < NPAD; k0 += 32) {
    v16h a = *(const v16h*)(&at[rA][0] + k0 + hi * 16);
    v16h b0 = *(const v16h*)(Vt + (vrow + rA)      * NPAD + k0 + hi * 16);
    v16h b1 = *(const v16h*)(Vt + (vrow + 16 + rA) * NPAD + k0 + hi * 16);
    o0 = __builtin_amdgcn_wmma_f32_16x16x32_f16(false, a, false, b0, (short)0, o0, false, false);
    o1 = __builtin_amdgcn_wmma_f32_16x16x32_f16(false, a, false, b1, (short)0, o1, false, false);
  }
#pragma unroll
  for (int r = 0; r < 8; r++) {
    int qt = qt0 + r + hi * 8;
    if (qt < NTOK) {
      size_t t = (size_t)w * NTOK + qt;
      O[t * CCH + head * HDIM + rA]      = (_Float16)o0[r];
      O[t * CCH + head * HDIM + 16 + rA] = (_Float16)o1[r];
    }
  }
}

// ---------------- proj GEMM + reverse shift scatter + residual -> x2 ----------
__global__ __launch_bounds__(256)
void proj_gemm_kernel(const _Float16* __restrict__ A, const _Float16* __restrict__ Wt,
                      const float* __restrict__ bias, const float* __restrict__ x,
                      float* __restrict__ x2) {
  const int NB = CCH / 64;
  int lane = threadIdx.x & 31, wid = threadIdx.x >> 5;
  long tile = (long)blockIdx.x * 8 + wid;
  if (tile >= (long)(TOKS / 64) * NB) return;
  int nt = (int)(tile % NB), mt = (int)(tile / NB);
  int rA = lane & 15, hi = lane >> 4;
  int row0 = mt * 64, col0 = nt * 64;
  v8f acc[4][4];
  gemm_core64(A, CCH, Wt, CCH, row0, col0, CCH, rA, hi, acc);
#pragma unroll
  for (int nb = 0; nb < 4; nb++) {
    int n = col0 + nb * 16 + rA;
    float bj = bias[n];
#pragma unroll
    for (int mb = 0; mb < 4; mb++) {
#pragma unroll
      for (int r = 0; r < 8; r++) {
        int m = row0 + mb * 16 + r + hi * 8;
        int w = m / NTOK, nn = m % NTOK;
        int dbl = w >> 6, hbl = (w >> 3) & 7, wbl = w & 7;
        int td = nn / 49, th = (nn / 7) % 7, tw = nn % 7;
        int sd = dbl * WIN_D + td, sh = hbl * WIN_H + th, sw = wbl * WIN_W + tw;
        int id = (sd + SSD) & 15;
        int ih = sh + SSH; if (ih >= HH) ih -= HH;
        int iw = sw + SSW; if (iw >= WW) iw -= WW;
        size_t img = ((size_t)(id * HH + ih) * WW + iw) * CCH + n;
        x2[img] = x[img] + acc[mb][nb][r] + bj;
      }
    }
  }
}

// ---------------- FC1 GEMM + exact GELU -> y1 f16 -----------------------------
__global__ __launch_bounds__(256)
void fc1_gemm_kernel(const _Float16* __restrict__ A, const _Float16* __restrict__ Wt,
                     const float* __restrict__ bias, _Float16* __restrict__ y1) {
  const int NB = HID / 64;
  int lane = threadIdx.x & 31, wid = threadIdx.x >> 5;
  long tile = (long)blockIdx.x * 8 + wid;
  if (tile >= (long)(TOKS / 64) * NB) return;
  int nt = (int)(tile % NB), mt = (int)(tile / NB);
  int rA = lane & 15, hi = lane >> 4;
  int row0 = mt * 64, col0 = nt * 64;
  v8f acc[4][4];
  gemm_core64(A, CCH, Wt, CCH, row0, col0, CCH, rA, hi, acc);
#pragma unroll
  for (int nb = 0; nb < 4; nb++) {
    int n = col0 + nb * 16 + rA;
    float bj = bias[n];
#pragma unroll
    for (int mb = 0; mb < 4; mb++) {
#pragma unroll
      for (int r = 0; r < 8; r++) {
        int m = row0 + mb * 16 + r + hi * 8;
        float v = acc[mb][nb][r] + bj;
        float ge = 0.5f * v * (1.0f + erff(v * 0.70710678118654752f));
        y1[(size_t)m * HID + n] = (_Float16)ge;
      }
    }
  }
}

// ---------------- FC2 GEMM + bias + residual -> out f32 -----------------------
__global__ __launch_bounds__(256)
void fc2_gemm_kernel(const _Float16* __restrict__ A, const _Float16* __restrict__ Wt,
                     const float* __restrict__ bias, const float* __restrict__ x2,
                     float* __restrict__ out) {
  const int NB = CCH / 64;
  int lane = threadIdx.x & 31, wid = threadIdx.x >> 5;
  long tile = (long)blockIdx.x * 8 + wid;
  if (tile >= (long)(TOKS / 64) * NB) return;
  int nt = (int)(tile % NB), mt = (int)(tile / NB);
  int rA = lane & 15, hi = lane >> 4;
  int row0 = mt * 64, col0 = nt * 64;
  v8f acc[4][4];
  gemm_core64(A, HID, Wt, HID, row0, col0, HID, rA, hi, acc);
#pragma unroll
  for (int nb = 0; nb < 4; nb++) {
    int n = col0 + nb * 16 + rA;
    float bj = bias[n];
#pragma unroll
    for (int mb = 0; mb < 4; mb++) {
#pragma unroll
      for (int r = 0; r < 8; r++) {
        int m = row0 + mb * 16 + r + hi * 8;
        size_t o = (size_t)m * CCH + n;
        out[o] = x2[o] + acc[mb][nb][r] + bj;
      }
    }
  }
}

// ---------------- host-side orchestration -------------------------------------
extern "C" void kernel_launch(void* const* d_in, const int* in_sizes, int n_in,
                              void* d_out, int out_size, void* d_ws, size_t ws_size,
                              hipStream_t stream) {
  const float* x       = (const float*)d_in[0];
  const float* n1g     = (const float*)d_in[1];
  const float* n1b     = (const float*)d_in[2];
  const float* qkv_w   = (const float*)d_in[3];
  const float* qkv_b   = (const float*)d_in[4];
  const float* rel_b   = (const float*)d_in[5];
  const float* proj_w  = (const float*)d_in[6];
  const float* proj_b  = (const float*)d_in[7];
  const float* n2g     = (const float*)d_in[8];
  const float* n2b     = (const float*)d_in[9];
  const float* fc1_w   = (const float*)d_in[10];
  const float* fc1_b   = (const float*)d_in[11];
  const float* fc2_w   = (const float*)d_in[12];
  const float* fc2_b   = (const float*)d_in[13];
  float* out = (float*)d_out;

  char* base = (char*)d_ws;
  size_t off = 0;
  auto carve = [&](size_t bytes) { char* p = base + off; off += (bytes + 255) & ~(size_t)255; return p; };

  const size_t WIN_BYTES = (size_t)TOKS * CCH * 2;                 // 25.7 MB
  const size_t QKV_BYTES = (size_t)NWIN * NHEAD * NPAD * HDIM * 2; // 27.3 MB each
  _Float16* win  = (_Float16*)carve(WIN_BYTES);
  _Float16* Qb   = (_Float16*)carve(QKV_BYTES);
  _Float16* Kb   = (_Float16*)carve(QKV_BYTES);
  _Float16* Vt   = (_Float16*)carve(QKV_BYTES);
  float*    x2   = (float*)   carve((size_t)TOKS * CCH * 4);       // 51.4 MB
  _Float16* y1   = (_Float16*)carve((size_t)TOKS * HID * 2);       // 102.8 MB
  _Float16* wq16 = (_Float16*)carve((size_t)768 * CCH * 2);
  _Float16* wp16 = (_Float16*)carve((size_t)CCH * CCH * 2);
  _Float16* w116 = (_Float16*)carve((size_t)HID * CCH * 2);
  _Float16* w216 = (_Float16*)carve((size_t)CCH * HID * 2);
  // dataflow-safe aliases: win dead after QKV gemm; Qb dead after attention
  _Float16* attn_out = win;
  _Float16* h2       = Qb;

  // 1) weights -> f16
  f32_to_f16_kernel<<<(768 * CCH + 255) / 256, 256, 0, stream>>>(qkv_w, wq16, 768 * CCH);
  f32_to_f16_kernel<<<(CCH * CCH + 255) / 256, 256, 0, stream>>>(proj_w, wp16, CCH * CCH);
  f32_to_f16_kernel<<<(HID * CCH + 255) / 256, 256, 0, stream>>>(fc1_w, w116, HID * CCH);
  f32_to_f16_kernel<<<(CCH * HID + 255) / 256, 256, 0, stream>>>(fc2_w, w216, CCH * HID);

  // 2) zero padded Q/K/Vt (contiguous carve; rounded sizes are 256B multiples)
  size_t qkv_words = (3 * QKV_BYTES) / 4;
  zero_kernel<<<2048, 256, 0, stream>>>((unsigned int*)Qb, qkv_words);

  // 3) LN1 + shift + partition
  ln1_shift_part_kernel<<<TOKS / 8, 256, 0, stream>>>(x, n1g, n1b, win);

  // 4) QKV GEMM: (784 * 12) 64x64 wave tiles, 8 waves/block
  qkv_gemm_kernel<<<(TOKS / 64) * (768 / 64) / 8, 256, 0, stream>>>(win, wq16, qkv_b, Qb, Kb, Vt);

  // 5) fused window attention
  attn_kernel<<<dim3(25, NHEAD, NWIN), 32, 0, stream>>>(Qb, Kb, Vt, rel_b, attn_out);

  // 6) proj GEMM + reverse shift + residual
  proj_gemm_kernel<<<(TOKS / 64) * (CCH / 64) / 8, 256, 0, stream>>>(attn_out, wp16, proj_b, x, x2);

  // 7) LN2
  ln2_kernel<<<TOKS / 8, 256, 0, stream>>>(x2, n2g, n2b, h2);

  // 8) FC1 + GELU
  fc1_gemm_kernel<<<(TOKS / 64) * (HID / 64) / 8, 256, 0, stream>>>(h2, w116, fc1_b, y1);

  // 9) FC2 + residual -> output
  fc2_gemm_kernel<<<(TOKS / 64) * (CCH / 64) / 8, 256, 0, stream>>>(y1, w216, fc2_b, x2, out);

  (void)in_sizes; (void)n_in; (void)out_size; (void)ws_size;
}
